// MinMaxAttention_38628935860713
// MI455X (gfx1250) — matrix operand order repaired and checked by reading, not compiled, not run on hardware
//
#include <hip/hip_runtime.h>
#include <hip/hip_bf16.h>

// ---------------------------------------------------------------------------
// Types
// ---------------------------------------------------------------------------
typedef __bf16 bf16;
typedef __attribute__((ext_vector_type(8)))  __bf16 v8bf;
typedef __attribute__((ext_vector_type(16))) __bf16 v16bf;
typedef __attribute__((ext_vector_type(8)))  float  v8f;

static __device__ inline v8f v8f_zero() {
    v8f z;
#pragma unroll
    for (int i = 0; i < 8; ++i) z[i] = 0.0f;
    return z;
}

static __device__ inline v8f wmma_bf16(v16bf a, v16bf b, v8f c) {
    // D = A(16x32 bf16) * B(32x16 bf16) + C(16x16 f32)
    return __builtin_amdgcn_wmma_f32_16x16x32_bf16(
        false, a, false, b, (short)0, c, false, false);
}

// Build a v16bf A/B fragment from two contiguous 8-element (16B) runs.
static __device__ inline v16bf ld_frag16(const bf16* p0, const bf16* p1) {
    v8bf lo = *(const v8bf*)p0;
    v8bf hi = *(const v8bf*)p1;
    v16bf r;
#pragma unroll
    for (int j = 0; j < 8; ++j) { r[j] = lo[j]; r[j + 8] = hi[j]; }
    return r;
}

// ---------------------------------------------------------------------------
// CDNA5 async global->LDS copy (GV mode): copies 32B per call site pair.
// INST_OFFSET is added to BOTH the LDS and the global address (ISA §10 async).
// Tracked by ASYNCcnt; must s_wait_asynccnt before consuming / barrier.
// ---------------------------------------------------------------------------
static __device__ inline void async_copy_b128(void* lds_dst, const void* gsrc) {
    const unsigned lds_addr = (unsigned)(uintptr_t)lds_dst;        // AS3 byte offset
    const unsigned long long gaddr = (unsigned long long)(uintptr_t)gsrc;
    asm volatile("global_load_async_to_lds_b128 %0, %1, off"
                 :
                 : "v"(lds_addr), "v"(gaddr)
                 : "memory");
}
static __device__ inline void async_copy_b128_off16(void* lds_dst, const void* gsrc) {
    const unsigned lds_addr = (unsigned)(uintptr_t)lds_dst;
    const unsigned long long gaddr = (unsigned long long)(uintptr_t)gsrc;
    asm volatile("global_load_async_to_lds_b128 %0, %1, off offset:16"
                 :
                 : "v"(lds_addr), "v"(gaddr)
                 : "memory");
}
static __device__ inline void wait_async_zero() {
    asm volatile("s_wait_asynccnt 0x0" ::: "memory");
}

// ---------------------------------------------------------------------------
// f32 -> bf16 convert
// ---------------------------------------------------------------------------
__global__ __launch_bounds__(256) void f32_to_bf16_kernel(const float* __restrict__ in,
                                                          bf16* __restrict__ out, int n) {
    int i = (blockIdx.x * 256 + threadIdx.x) * 8;
    if (i + 8 <= n) {
#pragma unroll
        for (int j = 0; j < 8; ++j) out[i + j] = (bf16)in[i + j];
    }
}

// ---------------------------------------------------------------------------
// bf16 WMMA GEMM:  C = act(A[MxK] * B[KxN])
//   BM=128 BN=128 BK=32, 256 threads = 8 waves (2x4), wave tile 64x32 (4x2 WMMA)
//   ACT: 0=none 1=silu 2=sigmoid ; OUTF32: 1 -> float out else bf16 out
// ---------------------------------------------------------------------------
template <int ACT, int OUTF32>
__global__ __launch_bounds__(256) void gemm_bf16_kernel(const bf16* __restrict__ A,
                                                        const bf16* __restrict__ B,
                                                        void* __restrict__ Cout,
                                                        int M, int N, int K) {
    __shared__ bf16 As[128][48];   // row-major, padded (96B rows, 16B aligned)
    __shared__ bf16 BsT[128][48];  // B transposed: [n][k], padded

    const int tid  = threadIdx.x;
    const int lane = tid & 31;
    const int wave = tid >> 5;
    const int wm   = wave >> 2;       // 0..1
    const int wn   = wave & 3;        // 0..3
    const int rowB = blockIdx.y * 128;
    const int colB = blockIdx.x * 128;

    const int r     = lane & 15;           // row (A) / col (B) within 16
    const int koff8 = (lane >> 4) * 8;     // A: K sub-offset 0 or 8
    const int kblk  = (lane >> 4) * 16;    // unused; B uses koff8 too

    const int arow = tid >> 1, ako = (tid & 1) * 16;        // A staging
    const int bkr  = tid >> 3, bno = (tid & 7) * 16;        // B staging

    v8f acc[4][2];
#pragma unroll
    for (int mi = 0; mi < 4; ++mi)
#pragma unroll
        for (int ni = 0; ni < 2; ++ni) acc[mi][ni] = v8f_zero();

    for (int kb = 0; kb < K; kb += 32) {
        // ---- stage A tile [128x32]: async global->LDS (32B / thread) ----
        {
            const bf16* g = &A[(size_t)(rowB + arow) * K + kb + ako];
            async_copy_b128(&As[arow][ako], g);        // bytes [0,16)
            async_copy_b128_off16(&As[arow][ako], g);  // bytes [16,32) both sides
        }
        // ---- stage B tile [32x128] transposed into BsT[n][k] ----
        {
            const bf16* g = &B[(size_t)(kb + bkr) * N + colB + bno];
            v8bf b0 = *(const v8bf*)g;
            v8bf b1 = *(const v8bf*)(g + 8);
#pragma unroll
            for (int j = 0; j < 8; ++j) {
                BsT[bno + j][bkr]     = b0[j];
                BsT[bno + 8 + j][bkr] = b1[j];
            }
        }
        if (kb + 32 < K) {  // CDNA5 global_prefetch_b8 of next tiles
            __builtin_prefetch(&A[(size_t)(rowB + arow) * K + kb + 32 + ako], 0, 0);
            __builtin_prefetch(&B[(size_t)(kb + 32 + bkr) * N + colB + bno], 0, 0);
        }
        wait_async_zero();
        __syncthreads();

        v16bf af[4];
#pragma unroll
        for (int mi = 0; mi < 4; ++mi) {
            const int rr = wm * 64 + mi * 16 + r;
            af[mi] = ld_frag16(&As[rr][koff8], &As[rr][16 + koff8]);
        }
        v16bf bfr[2];
#pragma unroll
        for (int ni = 0; ni < 2; ++ni) {
            const int cc = wn * 32 + ni * 16 + r;
            bfr[ni] = ld_frag16(&BsT[cc][koff8], &BsT[cc][16 + koff8]);
        }
#pragma unroll
        for (int mi = 0; mi < 4; ++mi)
#pragma unroll
            for (int ni = 0; ni < 2; ++ni)
                acc[mi][ni] = wmma_bf16(af[mi], bfr[ni], acc[mi][ni]);
        __syncthreads();
    }

    // ---- epilogue ----
    const int rowoff = (lane < 16) ? 0 : 8;
    const int ccol   = lane & 15;
#pragma unroll
    for (int mi = 0; mi < 4; ++mi)
#pragma unroll
        for (int ni = 0; ni < 2; ++ni)
#pragma unroll
            for (int i = 0; i < 8; ++i) {
                const size_t gr = (size_t)(rowB + wm * 64 + mi * 16 + i + rowoff);
                const size_t gc = (size_t)(colB + wn * 32 + ni * 16 + ccol);
                float v = acc[mi][ni][i];
                if (ACT == 1) v = v / (1.0f + expf(-v));         // silu
                else if (ACT == 2) v = 1.0f / (1.0f + expf(-v)); // sigmoid
                if (OUTF32) ((float*)Cout)[gr * N + gc] = v;
                else        ((bf16*)Cout)[gr * N + gc] = (bf16)v;
            }
}

// ---------------------------------------------------------------------------
// Blocked linear attention with ALiBi-style decay (lightning attention scan).
// grid = 32 (b*h chains), block = 256 threads (8 waves).
// Dynamic LDS layout (bytes):
//   Ks  [256][144] bf16  (k block, row-major)           73728
//   VsT [128][272] bf16  (v block, transposed [d][n])   69632
//   kvT [128][132] f32   (kv state, transposed [de][dk])67584
//   qks [8][16][272] bf16 (per-wave masked qk strip)    69632
// total = 280576 B  (<= 320KB WGP LDS)
// ---------------------------------------------------------------------------
#define ATTN_SMEM_BYTES 280576

__global__ __launch_bounds__(256) void attn_kernel(const bf16* __restrict__ qkv,
                                                   float* __restrict__ o) {
    extern __shared__ char smem_raw[];
    bf16*  Ks  = (bf16*)smem_raw;            // 256*144
    bf16*  VsT = Ks + 256 * 144;             // 128*272
    float* kvT = (float*)(VsT + 128 * 272);  // 128*132
    bf16*  qks = (bf16*)(kvT + 128 * 132);   // 8*16*272

    const int tid = threadIdx.x, lane = tid & 31, wave = tid >> 5;
    const int bh = blockIdx.x, bIdx = bh >> 4, h = bh & 15;

    // h=16 heads -> slope_i = 2^(-0.5*(i+1))
    const float slope     = exp2f(-0.5f * (float)(h + 1));
    const float blk_decay = expf(-slope * 256.0f);
    const int colQ = h * 384, colK = colQ + 128, colV = colQ + 256;

    for (int i = tid; i < 128 * 132; i += 256) kvT[i] = 0.0f;
    __syncthreads();

    const int r     = lane & 15;
    const int koff8 = (lane >> 4) * 8;
    const int kblk  = (lane >> 4) * 16;
    const int rowoff = (lane < 16) ? 0 : 8;

    for (int t = 0; t < 16; ++t) {
        const size_t rowBase = (size_t)bIdx * 4096 + (size_t)t * 256;

        // ---- stage Ks (async global->LDS) and VsT (register transpose) ----
#pragma unroll
        for (int it = 0; it < 8; ++it) {
            const int c  = tid + it * 256;      // 16-elem chunk id, 0..2047
            const int rw = c >> 3, ko = (c & 7) * 16;
            const bf16* gk = &qkv[(rowBase + rw) * 6144 + colK + ko];
            async_copy_b128(&Ks[rw * 144 + ko], gk);        // 16B
            async_copy_b128_off16(&Ks[rw * 144 + ko], gk);  // next 16B both sides
            const bf16* gv = &qkv[(rowBase + rw) * 6144 + colV + ko];
            v8bf v0 = *(const v8bf*)gv, v1 = *(const v8bf*)(gv + 8);
#pragma unroll
            for (int j = 0; j < 8; ++j) {
                VsT[(ko + j) * 272 + rw]     = v0[j];
                VsT[(ko + 8 + j) * 272 + rw] = v1[j];
            }
        }
        wait_async_zero();
        __syncthreads();

        // ============== Phase A: outputs, per wave 2 m-strips ==============
        for (int si = 0; si < 2; ++si) {
            const int m0 = (wave * 2 + si) * 16;

            // q fragments from global (raw + q_decay-scaled)
            v16bf aq[4], aqd[4];
            const float qd = expf(-slope * (float)(m0 + r + 1));
#pragma unroll
            for (int kk = 0; kk < 4; ++kk) {
                const bf16* g = &qkv[(rowBase + m0 + r) * 6144 + colQ + kk * 32 + koff8];
                v8bf lo = *(const v8bf*)g, hi = *(const v8bf*)(g + 16);
                v16bf a, ad;
#pragma unroll
                for (int j = 0; j < 8; ++j) {
                    a[j] = lo[j]; a[j + 8] = hi[j];
                    ad[j]     = (bf16)((float)lo[j] * qd);
                    ad[j + 8] = (bf16)((float)hi[j] * qd);
                }
                aq[kk] = a; aqd[kk] = ad;
            }

            // qk = qi @ ki^T  (two halves of n' to cap register pressure),
            // mask with causal decay, store bf16 to per-wave scratch
            bf16* myqk = qks + wave * 16 * 272;
            for (int half = 0; half < 2; ++half) {
                v8f qkacc[8];
#pragma unroll
                for (int nt = 0; nt < 8; ++nt) qkacc[nt] = v8f_zero();
#pragma unroll
                for (int kk = 0; kk < 4; ++kk) {
#pragma unroll
                    for (int nt = 0; nt < 8; ++nt) {
                        const int np = (half * 8 + nt) * 16 + r;  // lane's n' column
                        v16bf b = ld_frag16(&Ks[np * 144 + kk * 32 + koff8],
                                            &Ks[np * 144 + kk * 32 + koff8 + 16]);
                        qkacc[nt] = wmma_bf16(aq[kk], b, qkacc[nt]);
                    }
                }
#pragma unroll
                for (int nt = 0; nt < 8; ++nt) {
                    const int nn = (half * 8 + nt) * 16 + r;
#pragma unroll
                    for (int i = 0; i < 8; ++i) {
                        const int mloc = m0 + i + rowoff;
                        float v = qkacc[nt][i];
                        v = (mloc >= nn) ? v * expf(-slope * (float)(mloc - nn)) : 0.0f;
                        myqk[(i + rowoff) * 272 + nn] = (bf16)v;
                    }
                }
            }

            // out = (qi*q_decay) @ kv  +  qk_masked @ vi
            v8f oacc[8];
#pragma unroll
            for (int nt = 0; nt < 8; ++nt) oacc[nt] = v8f_zero();
            // kv part (K = 128 over dk), B from kvT[de][dk] (f32 -> bf16)
#pragma unroll
            for (int kk = 0; kk < 4; ++kk)
#pragma unroll
                for (int nt = 0; nt < 8; ++nt) {
                    const int de = nt * 16 + r;
                    v16bf b;
#pragma unroll
                    for (int j = 0; j < 16; ++j)
                        b[j] = (bf16)kvT[de * 132 + kk * 32 + koff8 + (j & 7) + (j >> 3) * 16];
                    oacc[nt] = wmma_bf16(aqd[kk], b, oacc[nt]);
                }
            // diag part (K = 256 over n'), A from qk scratch, B from VsT
#pragma unroll
            for (int kk2 = 0; kk2 < 8; ++kk2) {
                v16bf a2 = ld_frag16(&myqk[r * 272 + kk2 * 32 + koff8],
                                     &myqk[r * 272 + kk2 * 32 + 16 + koff8]);
#pragma unroll
                for (int nt = 0; nt < 8; ++nt) {
                    const int de = nt * 16 + r;
                    v16bf b = ld_frag16(&VsT[de * 272 + kk2 * 32 + koff8],
                                        &VsT[de * 272 + kk2 * 32 + koff8 + 16]);
                    oacc[nt] = wmma_bf16(a2, b, oacc[nt]);
                }
            }
            // write out f32, layout [b*n][h*d]
#pragma unroll
            for (int nt = 0; nt < 8; ++nt) {
                const int gc = h * 128 + nt * 16 + r;
#pragma unroll
                for (int i = 0; i < 8; ++i) {
                    const size_t gr = rowBase + (size_t)(m0 + i + rowoff);
                    o[gr * 2048 + gc] = oacc[nt][i];
                }
            }
        }
        __syncthreads();

        // ============== Phase B: kv = blk_decay*kv + (ki*k_decay)^T @ vi ====
        {
            const int dk0 = wave * 16;
            v8f kvacc[8];
#pragma unroll
            for (int nt = 0; nt < 8; ++nt) {
                const int de = nt * 16 + r;
#pragma unroll
                for (int i = 0; i < 8; ++i)
                    kvacc[nt][i] = blk_decay * kvT[de * 132 + dk0 + i + rowoff];
            }
#pragma unroll
            for (int kk2 = 0; kk2 < 8; ++kk2) {
                // A[dk][n] = ki[n][dk] * exp(-slope*(255-n))  (strided LDS reads)
                v16bf a3;
#pragma unroll
                for (int j = 0; j < 8; ++j) {
                    const int n0 = kk2 * 32 + koff8 + j;
                    const int n1 = kk2 * 32 + 16 + koff8 + j;
                    a3[j]     = (bf16)((float)Ks[n0 * 144 + dk0 + r] *
                                       expf(-slope * (float)(255 - n0)));
                    a3[j + 8] = (bf16)((float)Ks[n1 * 144 + dk0 + r] *
                                       expf(-slope * (float)(255 - n1)));
                }
#pragma unroll
                for (int nt = 0; nt < 8; ++nt) {
                    const int de = nt * 16 + r;
                    v16bf b = ld_frag16(&VsT[de * 272 + kk2 * 32 + koff8],
                                        &VsT[de * 272 + kk2 * 32 + koff8 + 16]);
                    kvacc[nt] = wmma_bf16(a3, b, kvacc[nt]);
                }
            }
            __syncthreads();  // all reads of old kvT complete before overwrite
#pragma unroll
            for (int nt = 0; nt < 8; ++nt) {
                const int de = nt * 16 + r;
#pragma unroll
                for (int i = 0; i < 8; ++i)
                    kvT[de * 132 + dk0 + i + rowoff] = kvacc[nt][i];
            }
        }
        __syncthreads();
    }
}

// ---------------------------------------------------------------------------
// RMS-norm + sigmoid-gate fusion:  og = bf16( o * rsqrt(mean(o^2)+eps) * w * g )
// ---------------------------------------------------------------------------
__global__ __launch_bounds__(256) void rmsnorm_gate_kernel(const float* __restrict__ o,
                                                           const bf16* __restrict__ g,
                                                           const float* __restrict__ nw,
                                                           bf16* __restrict__ og) {
    __shared__ float red[256];
    const size_t row = blockIdx.x;
    const float* orow = o + row * 2048;
    float s = 0.0f;
    for (int j = threadIdx.x; j < 2048; j += 256) { float v = orow[j]; s += v * v; }
    red[threadIdx.x] = s;
    __syncthreads();
    for (int off = 128; off > 0; off >>= 1) {
        if (threadIdx.x < off) red[threadIdx.x] += red[threadIdx.x + off];
        __syncthreads();
    }
    const float scale = rsqrtf(red[0] / 2048.0f + 1e-6f);
    for (int j = threadIdx.x; j < 2048; j += 256) {
        float v = orow[j] * scale * nw[j] * (float)g[row * 2048 + j];
        og[row * 2048 + j] = (bf16)v;
    }
}

// ---------------------------------------------------------------------------
// Host launcher
// ---------------------------------------------------------------------------
extern "C" void kernel_launch(void* const* d_in, const int* in_sizes, int n_in,
                              void* d_out, int out_size, void* d_ws, size_t ws_size,
                              hipStream_t stream) {
    const float* x      = (const float*)d_in[0];  // [2,4096,2048]
    const float* Wqkv   = (const float*)d_in[1];  // [2048,6144]
    const float* Wg     = (const float*)d_in[2];  // [2048,2048]
    const float* Wout   = (const float*)d_in[3];  // [2048,2048]
    const float* norm_w = (const float*)d_in[4];  // [2048]

    const int M = 8192, K = 2048, Nqkv = 6144, Nd = 2048;

    // workspace carve (256B aligned)
    char* ws = (char*)d_ws;
    auto carve = [&](size_t bytes) {
        char* p = ws;
        ws += (bytes + 255) & ~(size_t)255;
        return p;
    };
    bf16* xb    = (bf16*)carve((size_t)M * K * 2);
    bf16* wqkvb = (bf16*)carve((size_t)K * Nqkv * 2);
    bf16* wgb   = (bf16*)carve((size_t)K * Nd * 2);
    bf16* woutb = (bf16*)carve((size_t)K * Nd * 2);
    bf16* qkvb  = (bf16*)carve((size_t)M * Nqkv * 2);
    bf16* gb    = (bf16*)carve((size_t)M * Nd * 2);
    float* obuf = (float*)carve((size_t)M * Nd * 4);
    bf16* ogb   = (bf16*)carve((size_t)M * Nd * 2);

    // allow >64KB dynamic LDS for the attention kernel
    hipFuncSetAttribute((const void*)attn_kernel,
                        hipFuncAttributeMaxDynamicSharedMemorySize, ATTN_SMEM_BYTES);

    // 1) convert inputs to bf16
    f32_to_bf16_kernel<<<(M * K) / 2048, 256, 0, stream>>>(x, xb, M * K);
    f32_to_bf16_kernel<<<(K * Nqkv) / 2048, 256, 0, stream>>>(Wqkv, wqkvb, K * Nqkv);
    f32_to_bf16_kernel<<<(K * Nd) / 2048, 256, 0, stream>>>(Wg, wgb, K * Nd);
    f32_to_bf16_kernel<<<(K * Nd) / 2048, 256, 0, stream>>>(Wout, woutb, K * Nd);

    // 2) qkv = silu(x @ Wqkv)   3) g = sigmoid(x @ Wg)
    gemm_bf16_kernel<1, 0><<<dim3(Nqkv / 128, M / 128), 256, 0, stream>>>(
        xb, wqkvb, (void*)qkvb, M, Nqkv, K);
    gemm_bf16_kernel<2, 0><<<dim3(Nd / 128, M / 128), 256, 0, stream>>>(
        xb, wgb, (void*)gb, M, Nd, K);

    // 4) blocked decayed linear attention scan
    attn_kernel<<<32, 256, ATTN_SMEM_BYTES, stream>>>(qkvb, obuf);

    // 5) RMS norm + output gate
    rmsnorm_gate_kernel<<<M, 256, 0, stream>>>(obuf, gb, norm_w, ogb);

    // 6) out = og @ Wout  (f32 result)
    gemm_bf16_kernel<0, 1><<<dim3(Nd / 128, M / 128), 256, 0, stream>>>(
        ogb, woutb, d_out, M, Nd, K);
}
